// LoRAMoELayer_48576080118362
// MI455X (gfx1250) — matrix operand order repaired and checked by reading, not compiled, not run on hardware
//
#include <hip/hip_runtime.h>
#include <hip/hip_bf16.h>

typedef __attribute__((ext_vector_type(16))) _Float16 v16h;
typedef __attribute__((ext_vector_type(8)))  _Float16 v8h;
typedef __attribute__((ext_vector_type(8)))  float    v8f;

#define NUM_EXPERTS 8
#define RANK        16
#define IN_F        2048
#define OUT_F       2048
#define SCALING     2.0f
#define KAUG        (NUM_EXPERTS * RANK)   // 128 augmented K (u / lora_B)
#define KTOT        (IN_F + KAUG)          // 2176 total K for fused GEMM
#define PCOLS       144                    // 128 Ax + 8 logits + 8 pad

// ---- CDNA5 async global->LDS copy (16 bytes per lane), ASYNCcnt-tracked ----
__device__ __forceinline__ void async_cp16(unsigned lds_off, const _Float16* g) {
  asm volatile("global_load_async_to_lds_b128 %0, %1, off"
               :: "v"(lds_off), "v"((unsigned long long)(uintptr_t)g)
               : "memory");
}
#define ASYNC_WAIT0() asm volatile("s_wait_asynccnt 0x0" ::: "memory")
#define ASYNC_WAIT4() asm volatile("s_wait_asynccnt 0x4" ::: "memory")

__device__ __forceinline__ unsigned lds_addr(const void* p) {
  return (unsigned)(uintptr_t)p;  // low 32 bits of generic addr == LDS offset
}

// ---- fp32 -> f16 8-wide convert helper ----
__device__ __forceinline__ v8h cvt8(const float* s) {
  float4 f0 = *reinterpret_cast<const float4*>(s);
  float4 f1 = *reinterpret_cast<const float4*>(s + 4);
  v8h h;
  h[0] = (_Float16)f0.x; h[1] = (_Float16)f0.y;
  h[2] = (_Float16)f0.z; h[3] = (_Float16)f0.w;
  h[4] = (_Float16)f1.x; h[5] = (_Float16)f1.y;
  h[6] = (_Float16)f1.z; h[7] = (_Float16)f1.w;
  return h;
}

// ---------------------------------------------------------------------------
// Prep kernels (bandwidth-bound, run once): build f16 operands.
// ---------------------------------------------------------------------------
// xa[n, 0:2048] = f16(x[n,:]); cols 2048:2176 filled later by routing finish.
__global__ __launch_bounds__(256) void prep_xa(
    const float* __restrict__ x, _Float16* __restrict__ xa) {
  const int gid = blockIdx.x * 256 + threadIdx.x;
  const int n = gid >> 8, c8 = (gid & 255) * 8;
  *reinterpret_cast<v8h*>(xa + (size_t)n * KTOT + c8) =
      cvt8(x + (size_t)n * IN_F + c8);
}

// Wa[o, 0:2048] = f16(weight[o,:]); Wa[o, 2048+e*16+r] = f16(lora_B[e,o,r])
__global__ __launch_bounds__(256) void prep_wa(
    const float* __restrict__ weight, const float* __restrict__ lora_B,
    _Float16* __restrict__ wa) {
  const int gid = blockIdx.x * 256 + threadIdx.x;
  const int o = gid / (KTOT / 8), c8 = (gid % (KTOT / 8)) * 8;
  v8h h;
  if (c8 < IN_F) {
    h = cvt8(weight + (size_t)o * IN_F + c8);
  } else {
    const int k2 = c8 - IN_F;           // e*16 + r, r in {0,8}
    const int e = k2 >> 4, r = k2 & 15;
    h = cvt8(lora_B + ((size_t)e * OUT_F + o) * RANK + r);
  }
  *reinterpret_cast<v8h*>(wa + (size_t)o * KTOT + c8) = h;
}

// Br[0:128] = f16(lora_A flat); Br[128:136] = f16(router_w); Br[136:144] = 0
__global__ __launch_bounds__(256) void prep_br(
    const float* __restrict__ lora_A, const float* __restrict__ router_w,
    _Float16* __restrict__ br) {
  const int gid = blockIdx.x * 256 + threadIdx.x;
  const int r = gid >> 8, c8 = (gid & 255) * 8;
  v8h h;
  if (r < KAUG)                    h = cvt8(lora_A + (size_t)r * IN_F + c8);
  else if (r < KAUG + NUM_EXPERTS) h = cvt8(router_w + (size_t)(r - KAUG) * IN_F + c8);
  else { for (int j = 0; j < 8; ++j) h[j] = (_Float16)0.0f; }
  *reinterpret_cast<v8h*>(br + (size_t)r * IN_F + c8) = h;
}

// ---------------------------------------------------------------------------
// Kernel 1: P[N,144] = xa[:, 0:2048] @ Br^T   (f16 WMMA, f32 acc)
// Block: 128 tokens x 144 cols, 8 waves; wave = 16 rows x 9 col-tiles.
// Async global->LDS staging, single buffer.
// ---------------------------------------------------------------------------
__global__ __launch_bounds__(256) void lora_route_gemm(
    const _Float16* __restrict__ xa, const _Float16* __restrict__ br,
    float* __restrict__ P) {
  __shared__ __align__(64) _Float16 As[128][32];
  __shared__ __align__(64) _Float16 Bs[PCOLS][32];

  const int tid = threadIdx.x;
  const int n0  = blockIdx.x * 128;
  const int wid = tid >> 5, lane = tid & 31;
  const int l16 = lane & 15, hi = lane >> 4;
  const int sr = tid >> 1, cseg = (tid & 1) * 16;

  v8f acc[9];
#pragma unroll
  for (int i = 0; i < 9; ++i) acc[i] = (v8f)0.0f;

  const _Float16* aSrc = xa + (size_t)(n0 + sr) * KTOT + cseg;
  const _Float16* bSrc = br + (size_t)sr * IN_F + cseg;
  const unsigned la = lds_addr(&As[sr][cseg]);
  const unsigned lb = lds_addr(&Bs[sr][cseg]);
  // tail rows 128..143 staged by threads 0..63 (one 16B chunk each)
  const int trr = 128 + (tid >> 2), tseg = (tid & 3) * 8;
  const _Float16* tSrc = br + (size_t)trr * IN_F + tseg;
  const unsigned lt = lds_addr(&Bs[trr][tseg]);

  for (int kt = 0; kt < IN_F / 32; ++kt) {
    const int k0 = kt * 32;
    async_cp16(la,      aSrc + k0);
    async_cp16(la + 16, aSrc + k0 + 8);
    async_cp16(lb,      bSrc + k0);
    async_cp16(lb + 16, bSrc + k0 + 8);
    if (tid < 64) async_cp16(lt, tSrc + k0);
    ASYNC_WAIT0();
    __syncthreads();

    const int arow = wid * 16 + l16;
    v8h a0 = *reinterpret_cast<const v8h*>(&As[arow][hi ? 16 : 0]);
    v8h a1 = *reinterpret_cast<const v8h*>(&As[arow][hi ? 24 : 8]);
    v16h av;
#pragma unroll
    for (int j = 0; j < 8; ++j) { av[j] = a0[j]; av[j + 8] = a1[j]; }
#pragma unroll
    for (int ni = 0; ni < 9; ++ni) {
      v16h bv = *reinterpret_cast<const v16h*>(&Bs[ni * 16 + l16][hi ? 16 : 0]);
      acc[ni] = __builtin_amdgcn_wmma_f32_16x16x32_f16(
          false, av, false, bv, (short)0, acc[ni], false, false);
    }
    __syncthreads();
  }

#pragma unroll
  for (int ni = 0; ni < 9; ++ni) {
    const int col = ni * 16 + l16;
    const int rbase = n0 + wid * 16 + hi * 8;
#pragma unroll
    for (int v = 0; v < 8; ++v)
      P[(size_t)(rbase + v) * PCOLS + col] = acc[ni][v];
  }
}

// ---------------------------------------------------------------------------
// Kernel 2: top-2 routing -> write augmented operand into xa[:, 2048:2176]
// ---------------------------------------------------------------------------
__global__ __launch_bounds__(256) void lora_route_finish(
    const float* __restrict__ P, _Float16* __restrict__ xa, int Ntok) {
  const int n = blockIdx.x * blockDim.x + threadIdx.x;
  if (n >= Ntok) return;
  const float* p = P + (size_t)n * PCOLS;
  float l[NUM_EXPERTS];
#pragma unroll
  for (int e = 0; e < NUM_EXPERTS; ++e) l[e] = p[KAUG + e];
  int i0 = 0;
#pragma unroll
  for (int e = 1; e < NUM_EXPERTS; ++e) if (l[e] > l[i0]) i0 = e;
  int i1 = (i0 == 0) ? 1 : 0;
#pragma unroll
  for (int e = 0; e < NUM_EXPERTS; ++e)
    if (e != i0 && l[e] > l[i1]) i1 = e;
  const float z1 = __expf(l[i1] - l[i0]);
  const float inv = 1.0f / (1.0f + z1);
  const float w0 = SCALING * inv;
  const float w1 = SCALING * z1 * inv;
  _Float16* u = xa + (size_t)n * KTOT + IN_F;
#pragma unroll 4
  for (int j = 0; j < KAUG; ++j) {
    const int e = j >> 4;
    const float w = (e == i0) ? w0 : ((e == i1) ? w1 : 0.0f);
    u[j] = (_Float16)(w * p[j]);
  }
}

// ---------------------------------------------------------------------------
// Kernel 3: out[N,2048] = xa[N,2176] @ Wa^T  (base + expert path fused)
// Block 128x128, 8 waves, wave = 32x64 (2x4 frags, 8 WMMA/k-tile).
// Double-buffered LDS with async global->LDS; s_wait_asynccnt 0x4 keeps
// the next tile's 4 async ops in flight behind the WMMAs.
// ---------------------------------------------------------------------------
__global__ __launch_bounds__(256) void lora_moe_gemm(
    const _Float16* __restrict__ xa, const _Float16* __restrict__ wa,
    float* __restrict__ out) {
  __shared__ __align__(64) _Float16 As[2][128][32];
  __shared__ __align__(64) _Float16 Bs[2][128][32];

  const int tid = threadIdx.x;
  const int o0  = blockIdx.x * 128;
  const int n0  = blockIdx.y * 128;
  const int wid = tid >> 5, lane = tid & 31;
  const int l16 = lane & 15, hi = lane >> 4;
  const int wm = (wid >> 1) * 32, wn = (wid & 1) * 64;
  const int sr = tid >> 1, cseg = (tid & 1) * 16;
  const int NKT = KTOT / 32;  // 68

  v8f acc[2][4];
#pragma unroll
  for (int mi = 0; mi < 2; ++mi)
#pragma unroll
    for (int ni = 0; ni < 4; ++ni) acc[mi][ni] = (v8f)0.0f;

  const _Float16* aSrc = xa + (size_t)(n0 + sr) * KTOT + cseg;
  const _Float16* bSrc = wa + (size_t)(o0 + sr) * KTOT + cseg;
  const unsigned laBase[2] = {lds_addr(&As[0][sr][cseg]), lds_addr(&As[1][sr][cseg])};
  const unsigned lbBase[2] = {lds_addr(&Bs[0][sr][cseg]), lds_addr(&Bs[1][sr][cseg])};

  auto issue = [&](int kt, int buf) {
    const int k0 = kt * 32;
    async_cp16(laBase[buf],      aSrc + k0);
    async_cp16(laBase[buf] + 16, aSrc + k0 + 8);
    async_cp16(lbBase[buf],      bSrc + k0);
    async_cp16(lbBase[buf] + 16, bSrc + k0 + 8);
  };

  issue(0, 0);
  for (int kt = 0; kt < NKT; ++kt) {
    const int buf = kt & 1;
    if (kt + 1 < NKT) { issue(kt + 1, buf ^ 1); ASYNC_WAIT4(); }
    else              { ASYNC_WAIT0(); }
    __syncthreads();

    v16h av[2], bv[4];
#pragma unroll
    for (int mi = 0; mi < 2; ++mi) {
      const int arow = wm + mi * 16 + l16;
      v8h a0 = *reinterpret_cast<const v8h*>(&As[buf][arow][hi ? 16 : 0]);
      v8h a1 = *reinterpret_cast<const v8h*>(&As[buf][arow][hi ? 24 : 8]);
#pragma unroll
      for (int j = 0; j < 8; ++j) { av[mi][j] = a0[j]; av[mi][j + 8] = a1[j]; }
    }
#pragma unroll
    for (int ni = 0; ni < 4; ++ni)
      bv[ni] = *reinterpret_cast<const v16h*>(&Bs[buf][wn + ni * 16 + l16][hi ? 16 : 0]);
#pragma unroll
    for (int mi = 0; mi < 2; ++mi)
#pragma unroll
      for (int ni = 0; ni < 4; ++ni)
        acc[mi][ni] = __builtin_amdgcn_wmma_f32_16x16x32_f16(
            false, av[mi], false, bv[ni], (short)0, acc[mi][ni], false, false);
    __syncthreads();
  }

#pragma unroll
  for (int mi = 0; mi < 2; ++mi) {
    const int tb = n0 + wm + mi * 16 + hi * 8;
#pragma unroll
    for (int ni = 0; ni < 4; ++ni) {
      const int oc = o0 + wn + ni * 16 + l16;
#pragma unroll
      for (int v = 0; v < 8; ++v)
        out[(size_t)(tb + v) * OUT_F + oc] = acc[mi][ni][v];
    }
  }
}

extern "C" void kernel_launch(void* const* d_in, const int* in_sizes, int n_in,
                              void* d_out, int out_size, void* d_ws, size_t ws_size,
                              hipStream_t stream) {
  const float* x        = (const float*)d_in[0];
  const float* weight   = (const float*)d_in[1];
  const float* lora_A   = (const float*)d_in[2];
  const float* lora_B   = (const float*)d_in[3];
  const float* router_w = (const float*)d_in[4];
  float* out = (float*)d_out;

  const int Ntok = in_sizes[0] / IN_F;  // 8192

  // workspace layout (all f16 regions 16B-aligned)
  char* ws = (char*)d_ws;
  _Float16* xa = (_Float16*)ws;                       // [N, 2176] f16
  ws += (size_t)Ntok * KTOT * sizeof(_Float16);
  _Float16* wa = (_Float16*)ws;                       // [2048, 2176] f16
  ws += (size_t)OUT_F * KTOT * sizeof(_Float16);
  _Float16* br = (_Float16*)ws;                       // [144, 2048] f16
  ws += (size_t)PCOLS * IN_F * sizeof(_Float16);
  float* P = (float*)ws;                              // [N, 144] f32

  dim3 blk(256);
  prep_xa<<<dim3(Ntok), blk, 0, stream>>>(x, xa);
  prep_wa<<<dim3(OUT_F * (KTOT / 8) / 256), blk, 0, stream>>>(weight, lora_B, wa);
  prep_br<<<dim3(PCOLS), blk, 0, stream>>>(lora_A, router_w, br);
  lora_route_gemm<<<dim3(Ntok / 128), blk, 0, stream>>>(xa, br, P);
  lora_route_finish<<<dim3((Ntok + 255) / 256), blk, 0, stream>>>(P, xa, Ntok);
  lora_moe_gemm<<<dim3(OUT_F / 128, Ntok / 128), blk, 0, stream>>>(xa, wa, out);
}
